// TopologyGuidedDCN_22488448762523
// MI455X (gfx1250) — matrix operand order repaired and checked by reading, not compiled, not run on hardware
//
#include <hip/hip_runtime.h>
#include <hip/hip_bf16.h>

typedef __attribute__((ext_vector_type(16))) _Float16 v16h;
typedef __attribute__((ext_vector_type(8)))  _Float16 v8h;
typedef __attribute__((ext_vector_type(8)))  float    v8f;

static constexpr int Bn = 2;
static constexpr int Cch = 64;
static constexpr int Hh = 192;
static constexpr int Wt = 192;
static constexpr int HW = Hh * Wt;
static constexpr int FEATC = 121;   // 73 (pin) + 16 + 16 + 16
static constexpr int PCc = 73;
static constexpr int TPR = Wt / 16; // tiles per row = 12

// K ordering (ours, mirrored in B packing): k' = t*CPAD + c, CPAD = Cin
// rounded up to 8.  Per CDNA5 §7.12.2, an A fragment for a lane is exactly two
// runs of 8 consecutive K values -> two ds_load_b128 from a channel-last LDS
// halo: addr(t, m, c) = ((t/3)*18 + m + t%3)*CPAD + c.

union AU { v16h v; v8h h[2]; };

__device__ __forceinline__ float lrelu_f(float v) { return v >= 0.f ? v : 0.2f * v; }
__device__ __forceinline__ float sigmoid_f(float v) { return 1.f / (1.f + expf(-v)); }

// ============================================================================
// Weight repack: (Cout, Cin, taps) f32 -> f16 B-fragments in tap-major-K
// fragment order: [(nt*nchunks + ch)*32 + lane]*16 + j, K = ch*32 + (j<8 ? 8*half + j : 16 + 8*half + j-8)
// ============================================================================
__global__ void pack_wfrag_k(const float* __restrict__ w, int Cout, int Cin, int taps,
                             int Cpad, int nchunks, _Float16* __restrict__ frag) {
  int bid = blockIdx.x;               // = nt*nchunks + ch
  int ch  = bid % nchunks;
  int nt  = bid / nchunks;
  int lane = threadIdx.x;
  int half = lane >> 4;
  int n    = nt * 16 + (lane & 15);
  _Float16* dst = frag + ((size_t)bid * 32 + lane) * 16;
  #pragma unroll
  for (int j = 0; j < 16; ++j) {
    int k = ch * 32 + (j < 8 ? 8 * half + j : 16 + 8 * half + (j - 8));
    int t = k / Cpad, c = k % Cpad;
    float v = 0.f;
    if (t < taps && c < Cin && n < Cout)
      v = w[((size_t)n * Cin + c) * taps + t];
    dst[j] = (_Float16)v;
  }
}

// ============================================================================
// Prep: topo = sigmoid(pw(lrelu(dw3x3(x)))), geom_code = lrelu(1x1(geom)),
// feat[0:64]=x, feat[64]=topo, feat[65:73]=geom_code.
// ============================================================================
__global__ void prep_k(const float* __restrict__ x, const float* __restrict__ geom,
                       const float* __restrict__ w_dw, const float* __restrict__ w_pw,
                       const float* __restrict__ b_pw, const float* __restrict__ w_gc,
                       float* __restrict__ feat) {
  int idx = blockIdx.x * blockDim.x + threadIdx.x;
  if (idx >= Bn * HW) return;
  int b = idx / HW;
  int p = idx % HW;
  int y = p / Wt, xp = p % Wt;
  const float* xb = x + (size_t)b * Cch * HW;
  float* fb = feat + (size_t)b * FEATC * HW;
  float acc = 0.f;
  for (int c = 0; c < Cch; ++c) {
    float s = 0.f;
    #pragma unroll
    for (int t = 0; t < 9; ++t) {
      int yy = y + t / 3 - 1, xx = xp + t % 3 - 1;
      if (yy >= 0 && yy < Hh && xx >= 0 && xx < Wt)
        s += xb[(size_t)c * HW + yy * Wt + xx] * w_dw[c * 9 + t];
    }
    acc += lrelu_f(s) * w_pw[c];
    fb[(size_t)c * HW + p] = xb[(size_t)c * HW + p];
  }
  fb[(size_t)64 * HW + p] = sigmoid_f(acc + b_pw[0]);
  const float* gb = geom + (size_t)b * Cch * HW;
  for (int g = 0; g < 8; ++g) {
    float s = 0.f;
    for (int c = 0; c < Cch; ++c) s += gb[(size_t)c * HW + p] * w_gc[g * Cch + c];
    fb[(size_t)(65 + g) * HW + p] = lrelu_f(s);
  }
}

// ============================================================================
// Generic 3x3 conv: implicit GEMM, channel-last f16 halo in LDS, tap-major K.
// One wave per 16-pixel tile.  Inner loop: 2x ds_load_b128 (A) +
// NT x (2x global_load_b128 (B) + v_wmma_f32_16x16x32_f16).
// ============================================================================
template <int NT, int CPAD, bool LRELU>
__global__ void conv3x3_wmma_k(const float* __restrict__ in, int Cin, int CinStride,
                               const _Float16* __restrict__ wfrag, int nchunks,
                               int ldsHalves,
                               const float* __restrict__ bias, int Cout,
                               float* __restrict__ out, int outCoff, int outCstride) {
  extern __shared__ _Float16 lds[];
  const int lane = threadIdx.x;
  int bid = blockIdx.x;
  int x0 = (bid % TPR) * 16;
  int row = bid / TPR;
  int y = row % Hh;
  int b = row / Hh;

  // zero LDS (incl. channel padding + guard rows for padded tail chunks)
  v8h hz = {(_Float16)0.f, (_Float16)0.f, (_Float16)0.f, (_Float16)0.f,
            (_Float16)0.f, (_Float16)0.f, (_Float16)0.f, (_Float16)0.f};
  for (int i = lane * 8; i < ldsHalves; i += 256) *(v8h*)(lds + i) = hz;
  __syncthreads();

  // fill channel-last halo: lds[(rr*18+cc)*CPAD + c]
  for (int i = lane; i < Cin * 54; i += 32) {
    int c = i / 54;
    int rc = i % 54;
    int rr = rc / 18, cc = rc % 18;
    int yy = y + rr - 1, xx = x0 + cc - 1;
    if (yy >= 0 && yy < Hh && xx >= 0 && xx < Wt) {
      float v = in[(((size_t)b * CinStride + c) * Hh + yy) * Wt + xx];
      lds[rc * CPAD + c] = (_Float16)v;
    }
  }
  __syncthreads();

  const int half = lane >> 4;
  const int ln = lane & 15;
  v8f zero = {0.f, 0.f, 0.f, 0.f, 0.f, 0.f, 0.f, 0.f};
  v8f acc[NT];
  #pragma unroll
  for (int t = 0; t < NT; ++t) acc[t] = zero;

  for (int ch = 0; ch < nchunks; ++ch) {
    int kb0 = ch * 32 + half * 8;
    int kb1 = kb0 + 16;
    int t0 = kb0 / CPAD, c0 = kb0 % CPAD;   // 8-run never crosses a tap
    int t1 = kb1 / CPAD, c1 = kb1 % CPAD;
    int a0 = ((t0 / 3) * 18 + ln + (t0 % 3)) * CPAD + c0;
    int a1 = ((t1 / 3) * 18 + ln + (t1 % 3)) * CPAD + c1;
    AU au;
    au.h[0] = *(const v8h*)(lds + a0);
    au.h[1] = *(const v8h*)(lds + a1);
    #pragma unroll
    for (int nt = 0; nt < NT; ++nt) {
      v16h Bf = *(const v16h*)(wfrag + (((size_t)nt * nchunks + ch) * 32 + lane) * 16);
      acc[nt] = __builtin_amdgcn_wmma_f32_16x16x32_f16(false, au.v, false, Bf,
                                                       (short)0, acc[nt], false, false);
    }
  }

  #pragma unroll
  for (int nt = 0; nt < NT; ++nt) {
    #pragma unroll
    for (int r = 0; r < 8; ++r) {
      int mm = r + 8 * half;          // pixel within tile
      int n = nt * 16 + ln;           // output channel
      if (n < Cout) {
        float v = acc[nt][r] + bias[n];
        if (LRELU) v = lrelu_f(v);
        out[(((size_t)b * outCstride + outCoff + n) * Hh + y) * Wt + x0 + mm] = v;
      }
    }
  }
}

// ============================================================================
// 1x1 fuse conv (K=121 padded to 128 -> 73 out) + pin residual.
// ============================================================================
__global__ void fuse1x1_wmma_k(const float* __restrict__ feat,
                               const _Float16* __restrict__ wfrag,
                               const float* __restrict__ bias,
                               float* __restrict__ out) {
  __shared__ _Float16 lds[16 * 128];  // lds[m*128 + c]
  const int lane = threadIdx.x;
  int bid = blockIdx.x;
  int x0 = (bid % TPR) * 16;
  int row = bid / TPR;
  int y = row % Hh;
  int b = row / Hh;
  const float* fb = feat + (size_t)b * FEATC * HW;

  v8h hz = {(_Float16)0.f, (_Float16)0.f, (_Float16)0.f, (_Float16)0.f,
            (_Float16)0.f, (_Float16)0.f, (_Float16)0.f, (_Float16)0.f};
  for (int i = lane * 8; i < 16 * 128; i += 256) *(v8h*)(lds + i) = hz;
  __syncthreads();
  for (int i = lane; i < FEATC * 16; i += 32) {
    int c = i / 16, mm = i % 16;
    lds[mm * 128 + c] = (_Float16)fb[(size_t)c * HW + (size_t)y * Wt + x0 + mm];
  }
  __syncthreads();

  const int half = lane >> 4;
  const int ln = lane & 15;
  v8f zero = {0.f, 0.f, 0.f, 0.f, 0.f, 0.f, 0.f, 0.f};
  v8f acc[5];
  #pragma unroll
  for (int t = 0; t < 5; ++t) acc[t] = zero;

  const int nchunks = 4;   // 128 / 32
  for (int ch = 0; ch < nchunks; ++ch) {
    int a0 = ln * 128 + ch * 32 + half * 8;
    AU au;
    au.h[0] = *(const v8h*)(lds + a0);
    au.h[1] = *(const v8h*)(lds + a0 + 16);
    #pragma unroll
    for (int nt = 0; nt < 5; ++nt) {
      v16h Bf = *(const v16h*)(wfrag + (((size_t)nt * nchunks + ch) * 32 + lane) * 16);
      acc[nt] = __builtin_amdgcn_wmma_f32_16x16x32_f16(false, au.v, false, Bf,
                                                       (short)0, acc[nt], false, false);
    }
  }

  #pragma unroll
  for (int nt = 0; nt < 5; ++nt) {
    #pragma unroll
    for (int r = 0; r < 8; ++r) {
      int mm = r + 8 * half;
      int n = nt * 16 + ln;
      if (n < PCc) {
        size_t pp = (size_t)y * Wt + x0 + mm;
        out[((size_t)b * PCc + n) * HW + pp] = acc[nt][r] + bias[n] + fb[(size_t)n * HW + pp];
      }
    }
  }
}

// ============================================================================
// Deformable conv.  Phase 1: bilinear coords/weights (mask+validity folded)
// for 16 px x 9 taps -> LDS.  Phase 2: materialize val[m][t][c] f16 in LDS
// (the gather).  Phase 3: dense GEMM, K = t*64+c = 576, 18 chunks x 4 N-tiles.
// out = x + gamma * (dcn + bias)
// ============================================================================
__global__ void dcn_wmma_k(const float* __restrict__ x, const float* __restrict__ om,
                           const _Float16* __restrict__ wfrag,
                           const float* __restrict__ b_dcn,
                           const float* __restrict__ gamma,
                           float* __restrict__ out) {
  __shared__ int   s_y0[144], s_y1[144], s_x0[144], s_x1[144];
  __shared__ float s_w[144][4];
  __shared__ _Float16 sval[16 * 576];   // sval[m*576 + t*64 + c]
  const int lane = threadIdx.x;
  int bid = blockIdx.x;
  int x0p = (bid % TPR) * 16;
  int row = bid / TPR;
  int y = row % Hh;
  int b = row / Hh;
  const float* omb = om + (size_t)b * 27 * HW;

  for (int i = lane; i < 144; i += 32) {
    int mi = i / 9, t = i % 9;
    int xx = x0p + mi;
    size_t pix = (size_t)y * Wt + xx;
    float offy = omb[(size_t)(2 * t) * HW + pix];
    float offx = omb[(size_t)(2 * t + 1) * HW + pix];
    float msk  = sigmoid_f(omb[(size_t)(18 + t) * HW + pix]);
    float py = offy + (float)(t / 3) + (float)y - 1.f;
    float px = offx + (float)(t % 3) + (float)xx - 1.f;
    float fy = floorf(py), fx = floorf(px);
    float ly = py - fy, lx = px - fx;
    int y0 = (int)fy, xq = (int)fx;
    bool vy0 = (y0 >= 0) && (y0 < Hh);
    bool vy1 = (y0 + 1 >= 0) && (y0 + 1 < Hh);
    bool vx0 = (xq >= 0) && (xq < Wt);
    bool vx1 = (xq + 1 >= 0) && (xq + 1 < Wt);
    s_w[i][0] = (1.f - ly) * (1.f - lx) * msk * ((vy0 && vx0) ? 1.f : 0.f);
    s_w[i][1] = (1.f - ly) * lx        * msk * ((vy0 && vx1) ? 1.f : 0.f);
    s_w[i][2] = ly * (1.f - lx)        * msk * ((vy1 && vx0) ? 1.f : 0.f);
    s_w[i][3] = ly * lx                * msk * ((vy1 && vx1) ? 1.f : 0.f);
    s_y0[i] = min(max(y0, 0), Hh - 1);
    s_y1[i] = min(max(y0 + 1, 0), Hh - 1);
    s_x0[i] = min(max(xq, 0), Wt - 1);
    s_x1[i] = min(max(xq + 1, 0), Wt - 1);
  }
  __syncthreads();

  const float* xb = x + (size_t)b * Cch * HW;
  for (int i = lane; i < 16 * 576; i += 32) {
    int c  = i & 63;
    int tt = (i >> 6) % 9;
    int m  = i / 576;
    int e  = m * 9 + tt;
    const float* xc = xb + (size_t)c * HW;
    float v = s_w[e][0] * xc[s_y0[e] * Wt + s_x0[e]]
            + s_w[e][1] * xc[s_y0[e] * Wt + s_x1[e]]
            + s_w[e][2] * xc[s_y1[e] * Wt + s_x0[e]]
            + s_w[e][3] * xc[s_y1[e] * Wt + s_x1[e]];
    sval[i] = (_Float16)v;
  }
  __syncthreads();

  const int half = lane >> 4;
  const int ln = lane & 15;
  v8f zero = {0.f, 0.f, 0.f, 0.f, 0.f, 0.f, 0.f, 0.f};
  v8f acc[4];
  #pragma unroll
  for (int t = 0; t < 4; ++t) acc[t] = zero;

  const int nchunks = 18;  // 576 / 32, exact
  for (int ch = 0; ch < nchunks; ++ch) {
    int a0 = ln * 576 + ch * 32 + half * 8;
    AU au;
    au.h[0] = *(const v8h*)(sval + a0);
    au.h[1] = *(const v8h*)(sval + a0 + 16);
    #pragma unroll
    for (int nt = 0; nt < 4; ++nt) {
      v16h Bf = *(const v16h*)(wfrag + (((size_t)nt * nchunks + ch) * 32 + lane) * 16);
      acc[nt] = __builtin_amdgcn_wmma_f32_16x16x32_f16(false, au.v, false, Bf,
                                                       (short)0, acc[nt], false, false);
    }
  }

  #pragma unroll
  for (int nt = 0; nt < 4; ++nt) {
    #pragma unroll
    for (int r = 0; r < 8; ++r) {
      int mm = r + 8 * half;
      int n = nt * 16 + ln;
      size_t p = ((size_t)(b * Cch + n) * Hh + y) * Wt + x0p + mm;
      out[p] = x[p] + gamma[n] * (acc[nt][r] + b_dcn[n]);
    }
  }
}

// ============================================================================
// Host launcher
// ============================================================================
extern "C" void kernel_launch(void* const* d_in, const int* in_sizes, int n_in,
                              void* d_out, int out_size, void* d_ws, size_t ws_size,
                              hipStream_t stream) {
  const float* x      = (const float*)d_in[0];
  const float* geom   = (const float*)d_in[1];
  const float* w_gc   = (const float*)d_in[2];
  const float* w_dw   = (const float*)d_in[3];
  const float* w_pw   = (const float*)d_in[4];
  const float* b_pw   = (const float*)d_in[5];
  const float* w_rdb1 = (const float*)d_in[6];
  const float* b_rdb1 = (const float*)d_in[7];
  const float* w_rdb2 = (const float*)d_in[8];
  const float* b_rdb2 = (const float*)d_in[9];
  const float* w_rdb3 = (const float*)d_in[10];
  const float* b_rdb3 = (const float*)d_in[11];
  const float* w_fuse = (const float*)d_in[12];
  const float* b_fuse = (const float*)d_in[13];
  const float* w_lh1  = (const float*)d_in[14];
  const float* b_lh1  = (const float*)d_in[15];
  const float* w_lh2  = (const float*)d_in[16];
  const float* b_lh2  = (const float*)d_in[17];
  const float* w_dcn  = (const float*)d_in[18];
  const float* b_dcn  = (const float*)d_in[19];
  const float* gamma  = (const float*)d_in[20];
  float* out = (float*)d_out;

  char* ws = (char*)d_ws;
  size_t off = 0;
  auto alloc = [&](size_t bytes) -> void* {
    void* p = ws + off;
    off = (off + bytes + 255) & ~(size_t)255;
    return p;
  };
  float* feat = (float*)alloc((size_t)Bn * FEATC * HW * 4);
  float* path = (float*)alloc((size_t)Bn * PCc * HW * 4);
  float* hid  = (float*)alloc((size_t)Bn * 32 * HW * 4);
  float* omb  = (float*)alloc((size_t)Bn * 27 * HW * 4);

  // CPAD = Cin rounded up to 8; nchunks covers max real k' = 8*CPAD + Cin - 1
  const int nc_rdb1 = 23;  // CPAD 80,  k'max 712
  const int nc_rdb2 = 27;  // CPAD 96,  k'max 856
  const int nc_rdb3 = 32;  // CPAD 112, k'max 1000
  const int nc_fuse = 4;   // CPAD 128 (1x1)
  const int nc_lh1  = 23;  // CPAD 80
  const int nc_lh2  = 9;   // CPAD 32
  const int nc_dcn  = 18;  // CPAD 64

  // LDS halves incl. guard rows: ((tmax/3)*18 + 15 + tmax%3 + 1) * CPAD
  const int lh_rdb1 = 70 * 80;    // tmax 9
  const int lh_rdb2 = 54 * 96;    // tmax 8
  const int lh_rdb3 = 70 * 112;   // tmax 9
  const int lh_lh1  = 70 * 80;
  const int lh_lh2  = 54 * 32;    // tmax 8

  _Float16* f_rdb1 = (_Float16*)alloc((size_t)1 * nc_rdb1 * 512 * 2);
  _Float16* f_rdb2 = (_Float16*)alloc((size_t)1 * nc_rdb2 * 512 * 2);
  _Float16* f_rdb3 = (_Float16*)alloc((size_t)1 * nc_rdb3 * 512 * 2);
  _Float16* f_fuse = (_Float16*)alloc((size_t)5 * nc_fuse * 512 * 2);
  _Float16* f_lh1  = (_Float16*)alloc((size_t)2 * nc_lh1 * 512 * 2);
  _Float16* f_lh2  = (_Float16*)alloc((size_t)2 * nc_lh2 * 512 * 2);
  _Float16* f_dcn  = (_Float16*)alloc((size_t)4 * nc_dcn * 512 * 2);

  pack_wfrag_k<<<1 * nc_rdb1, 32, 0, stream>>>(w_rdb1, 16, 73, 9, 80, nc_rdb1, f_rdb1);
  pack_wfrag_k<<<1 * nc_rdb2, 32, 0, stream>>>(w_rdb2, 16, 89, 9, 96, nc_rdb2, f_rdb2);
  pack_wfrag_k<<<1 * nc_rdb3, 32, 0, stream>>>(w_rdb3, 16, 105, 9, 112, nc_rdb3, f_rdb3);
  pack_wfrag_k<<<5 * nc_fuse, 32, 0, stream>>>(w_fuse, 73, 121, 1, 128, nc_fuse, f_fuse);
  pack_wfrag_k<<<2 * nc_lh1, 32, 0, stream>>>(w_lh1, 32, 73, 9, 80, nc_lh1, f_lh1);
  pack_wfrag_k<<<2 * nc_lh2, 32, 0, stream>>>(w_lh2, 27, 32, 9, 32, nc_lh2, f_lh2);
  pack_wfrag_k<<<4 * nc_dcn, 32, 0, stream>>>(w_dcn, 64, 64, 9, 64, nc_dcn, f_dcn);

  prep_k<<<(Bn * HW + 255) / 256, 256, 0, stream>>>(x, geom, w_dw, w_pw, b_pw, w_gc, feat);

  const int nTiles = Bn * Hh * TPR;  // 4608 waves

  conv3x3_wmma_k<1, 80, true><<<nTiles, 32, lh_rdb1 * 2, stream>>>(
      feat, 73, FEATC, f_rdb1, nc_rdb1, lh_rdb1, b_rdb1, 16, feat, 73, FEATC);
  conv3x3_wmma_k<1, 96, true><<<nTiles, 32, lh_rdb2 * 2, stream>>>(
      feat, 89, FEATC, f_rdb2, nc_rdb2, lh_rdb2, b_rdb2, 16, feat, 89, FEATC);
  conv3x3_wmma_k<1, 112, true><<<nTiles, 32, lh_rdb3 * 2, stream>>>(
      feat, 105, FEATC, f_rdb3, nc_rdb3, lh_rdb3, b_rdb3, 16, feat, 105, FEATC);

  fuse1x1_wmma_k<<<nTiles, 32, 0, stream>>>(feat, f_fuse, b_fuse, path);

  conv3x3_wmma_k<2, 80, true><<<nTiles, 32, lh_lh1 * 2, stream>>>(
      path, 73, PCc, f_lh1, nc_lh1, lh_lh1, b_lh1, 32, hid, 0, 32);
  conv3x3_wmma_k<2, 32, false><<<nTiles, 32, lh_lh2 * 2, stream>>>(
      hid, 32, 32, f_lh2, nc_lh2, lh_lh2, b_lh2, 27, omb, 0, 27);

  dcn_wmma_k<<<nTiles, 32, 0, stream>>>(x, omb, f_dcn, b_dcn, gamma, out);
}
